// MPNNBlock_11330123727326
// MI455X (gfx1250) — matrix operand order repaired and checked by reading, compiled, not run on hardware
//
#include <hip/hip_runtime.h>
#include <cstdint>

typedef __attribute__((ext_vector_type(16))) _Float16 v16h;
typedef __attribute__((ext_vector_type(8)))  float    v8f;
typedef _Float16 h16;

#define NB   16     // batch
#define NN   128    // nodes
#define HD   64     // hidden
#define ED   32     // edge dim
#define RTOT 2048   // NB*NN rows

// ---------------------------------------------------------------------------
// 16x16-tile WMMA GEMM, one wave32 per block; K, LDA, LDD compile-time so the
// K-loop fully unrolls and all addressing folds into immediate offsets.
//   D[m][n] (+bias[n]) = sum_k A[m][k] * B[k][n]
// A:  f32 row-major (LDA), per-batch stride aBatch
// Wp: f16, N x K row-major per batch (B[k][n] = Wp[n*K+k]) -> per-lane B
//     fragments are 2 contiguous 16B loads
// grid = (ntiles, mtiles, batch). K multiple of 32.
// Fragment packing: contiguous k = 16*kh+i in A and B (16-bit A 16x32; B mirrored; C/D f32).
// ---------------------------------------------------------------------------
template <int K, int LDA, int LDD>
__global__ __launch_bounds__(32)
void wmma_gemm_k(const float* __restrict__ A, int64_t aBatch,
                 const h16* __restrict__ Wp, int64_t wBatch,
                 const float* __restrict__ bias,
                 float* __restrict__ D, int64_t dBatch)
{
  const int nt = blockIdx.x, mt = blockIdx.y, b = blockIdx.z;
  const int lane = threadIdx.x & 31;
  const int lr   = lane & 15;   // A row / B col / D col
  const int kh   = lane >> 4;   // K-half selector
  const int n0   = nt * 16;

  const float* Ab = A + (int64_t)b * aBatch + (int64_t)(mt * 16 + lr) * LDA;
  const h16*   Wb = Wp + (int64_t)b * wBatch + (int64_t)(n0 + lr) * K;
  float*       Db = D + (int64_t)b * dBatch
                      + (int64_t)(mt * 16 + 8 * kh) * LDD + (n0 + lr);

  v8f acc = {};
#pragma unroll
  for (int k0 = 0; k0 < K; k0 += 32) {
    // A fragment: elems 0..15 -> k = k0+16*kh+i (contiguous, matches B)
    v16h af;
    {
      const float* a0 = Ab + k0 + 16 * kh;
      const float* a1 = Ab + k0 + 16 * kh + 8;
#pragma unroll
      for (int i = 0; i < 8; ++i) af[i] = (h16)a0[i];
#pragma unroll
      for (int i = 0; i < 8; ++i) af[8 + i] = (h16)a1[i];
    }
    // B fragment: lane half holds K = 16*kh .. 16*kh+15 for column n0+lr
    v16h bf;
    {
      const h16* wp = Wb + k0 + 16 * kh;
#pragma unroll
      for (int p = 0; p < 16; ++p) bf[p] = wp[p];
    }
    acc = __builtin_amdgcn_wmma_f32_16x16x32_f16(
        false, af, false, bf, (short)0, acc, false, false);
  }

  const float bv = bias ? bias[n0 + lr] : 0.0f;
  // C/D layout: lane row m = mt*16 + r + 8*kh ; compile-time LDD -> immediates
#pragma unroll
  for (int r = 0; r < 8; ++r) Db[r * LDD] = acc[r] + bv;
}

// ---------------------------------------------------------------------------
// ehat[r][k] = sum_j adj[r][j] * e[r][j][k];  deg[r] = sum_j adj[r][j]
// One block per row r (b*NN+i); the single streaming pass over the 33.5MB e.
// ---------------------------------------------------------------------------
__global__ __launch_bounds__(256)
void ehat_deg_k(const float* __restrict__ e, const float* __restrict__ adj,
                float* __restrict__ ehat, float* __restrict__ deg)
{
  const int r = blockIdx.x;
  __shared__ float sadj[NN];
  __shared__ float part[256];
  const int t = threadIdx.x;
  if (t < NN) sadj[t] = adj[(int64_t)r * NN + t];
  __syncthreads();
  const int k = t & (ED - 1), jg = t >> 5;       // 8 j-groups x 32 k
  const float* er = e + (int64_t)r * NN * ED;
  float acc = 0.0f;
  for (int j = jg; j < NN; j += 8) acc += sadj[j] * er[j * ED + k];
  part[t] = acc;
  __syncthreads();
  if (t < ED) {
    float s = 0.0f;
#pragma unroll
    for (int g = 0; g < 8; ++g) s += part[g * 32 + t];
    ehat[(int64_t)r * ED + t] = s;
    if (t == 0) {
      float d = 0.0f;
      for (int j = 0; j < NN; ++j) d += sadj[j];
      deg[r] = d;
    }
  }
}

// f32 -> f16 weight repack: dst[n*K+k] = src[n*rowStride + colOff + k]
__global__ void pack_f16_k(const float* __restrict__ src, int rowStride,
                           int colOff, h16* __restrict__ dst, int cnt, int K)
{
  int i = blockIdx.x * blockDim.x + threadIdx.x;
  if (i < cnt) dst[i] = (h16)src[(i / K) * rowStride + colOff + (i % K)];
}

// t1T[b][n][j] = (f16) P[(b*NN+j)*320 + n]  -- K-contiguous B operand for adj@t1
__global__ __launch_bounds__(256)
void pack_t1T_k(const float* __restrict__ P, h16* __restrict__ t1T)
{
  int i = blockIdx.x * blockDim.x + threadIdx.x;   // NB*HD*NN = 131072
  int b = i >> 13;                                  // / (HD*NN)
  int rem = i & 8191;
  int n = rem >> 7;                                 // / NN
  int j = rem & (NN - 1);
  t1T[i] = (h16)P[((b << 7) + j) * 320 + n];
}

// bias for the fused [t1|t2|gi] GEMM: zeros for first 128 cols, then bih
__global__ void bias320_k(const float* __restrict__ bih, float* __restrict__ dst)
{
  int i = blockIdx.x * blockDim.x + threadIdx.x;
  if (i < 320) dst[i] = (i < 128) ? 0.0f : bih[i - 128];
}

// m = (S + deg*(t2+bm) + Q) / N     (t2 = P[:,64:128])
__global__ __launch_bounds__(256)
void ew_m_k(const float* __restrict__ S, const float* __restrict__ P,
            const float* __restrict__ Q, const float* __restrict__ deg,
            const float* __restrict__ bm, float* __restrict__ mb)
{
  int idx = blockIdx.x * blockDim.x + threadIdx.x;
  int r = idx >> 6, n = idx & 63;
  float t2 = P[r * 320 + 64 + n];
  mb[idx] = (S[idx] + deg[r] * (t2 + bm[n]) + Q[idx]) * (1.0f / (float)NN);
}

// GRU combine: gi = P[:,128:320] (bih folded in), gh has bhh folded in.
__global__ __launch_bounds__(256)
void ew_gru_k(const float* __restrict__ P, const float* __restrict__ gh,
              const float* __restrict__ mb, float* __restrict__ hout)
{
  int idx = blockIdx.x * blockDim.x + threadIdx.x;
  int r = idx >> 6, n = idx & 63;
  float gr = P[r * 320 + 128 + n];
  float gz = P[r * 320 + 192 + n];
  float gn = P[r * 320 + 256 + n];
  float hr = gh[r * 192 + n];
  float hz = gh[r * 192 + 64 + n];
  float hn = gh[r * 192 + 128 + n];
  float rg = 1.0f / (1.0f + __expf(-(gr + hr)));
  float z  = 1.0f / (1.0f + __expf(-(gz + hz)));
  float nn = tanhf(gn + rg * hn);
  hout[idx] = (1.0f - z) * nn + z * mb[idx];
}

__global__ __launch_bounds__(256)
void add_k(const float* __restrict__ a, const float* __restrict__ b,
           float* __restrict__ o)
{
  int i = blockIdx.x * blockDim.x + threadIdx.x;
  o[i] = a[i] + b[i];
}

// ---------------------------------------------------------------------------
static void run_layer(const float* hcur, float* hout, const float* adj,
                      const float* bm, const float* bhh,
                      const h16* packA, const float* bias320,
                      const h16* packW3, const h16* packWhh,
                      const float* ehat, const float* deg,
                      float* P, float* Q, float* S, float* mb, float* gh,
                      h16* t1T, hipStream_t stream)
{
  // P = hcur @ [W1;W2;Wih]^T + [0|0|bih]   (2048 x 320, K=64)
  wmma_gemm_k<64, 64, 320><<<dim3(20, 128, 1), 32, 0, stream>>>(
      hcur, 0, packA, 0, bias320, P, 0);
  // Q = ehat @ W3^T                         (2048 x 64, K=32)
  wmma_gemm_k<32, 32, 64><<<dim3(4, 128, 1), 32, 0, stream>>>(
      ehat, 0, packW3, 0, nullptr, Q, 0);
  // repack t1 = P[:,0:64] transposed to f16 (per-batch N x K weight layout)
  pack_t1T_k<<<NB * HD * NN / 256, 256, 0, stream>>>(P, t1T);
  // S[b] = adj[b] @ t1[b]                   (batched, 128 x 64, K=128)
  wmma_gemm_k<128, 128, 64><<<dim3(4, 8, NB), 32, 0, stream>>>(
      adj, (int64_t)NN * NN, t1T, (int64_t)HD * NN, nullptr,
      S, (int64_t)NN * HD);
  ew_m_k<<<RTOT * HD / 256, 256, 0, stream>>>(S, P, Q, deg, bm, mb);
  // gh = m @ Whh^T + bhh                    (2048 x 192, K=64)
  wmma_gemm_k<64, 64, 192><<<dim3(12, 128, 1), 32, 0, stream>>>(
      mb, 0, packWhh, 0, bhh, gh, 0);
  ew_gru_k<<<RTOT * HD / 256, 256, 0, stream>>>(P, gh, mb, hout);
}

extern "C" void kernel_launch(void* const* d_in, const int* in_sizes, int n_in,
                              void* d_out, int out_size, void* d_ws, size_t ws_size,
                              hipStream_t stream)
{
  (void)in_sizes; (void)n_in; (void)out_size; (void)ws_size;
  const float* h    = (const float*)d_in[0];
  const float* e    = (const float*)d_in[1];
  const float* adj  = (const float*)d_in[2];
  const float* Wm1  = (const float*)d_in[3];
  const float* bm1  = (const float*)d_in[4];
  const float* Wih1 = (const float*)d_in[5];
  const float* Whh1 = (const float*)d_in[6];
  const float* bih1 = (const float*)d_in[7];
  const float* bhh1 = (const float*)d_in[8];
  const float* Wm2  = (const float*)d_in[9];
  const float* bm2  = (const float*)d_in[10];
  const float* Wih2 = (const float*)d_in[11];
  const float* Whh2 = (const float*)d_in[12];
  const float* bih2 = (const float*)d_in[13];
  const float* bhh2 = (const float*)d_in[14];
  float* out = (float*)d_out;

  // ---- workspace carve-up (~7.6 MB, 256B-aligned chunks) ----
  char* ws = (char*)d_ws;
  size_t off = 0;
  auto carve = [&](size_t bytes) {
    void* p = ws + off;
    off = (off + bytes + 255) & ~(size_t)255;
    return p;
  };
  float* ehat   = (float*)carve((size_t)RTOT * ED * 4);
  float* deg    = (float*)carve((size_t)RTOT * 4);
  float* P      = (float*)carve((size_t)RTOT * 320 * 4);
  float* Q      = (float*)carve((size_t)RTOT * HD * 4);
  float* S      = (float*)carve((size_t)RTOT * HD * 4);
  float* mb     = (float*)carve((size_t)RTOT * HD * 4);
  float* gh     = (float*)carve((size_t)RTOT * 192 * 4);
  float* h1     = (float*)carve((size_t)RTOT * HD * 4);
  float* h2     = (float*)carve((size_t)RTOT * HD * 4);
  h16*   t1T    = (h16*)carve((size_t)NB * HD * NN * 2);
  h16*   packA1 = (h16*)carve((size_t)320 * HD * 2);
  h16*   packA2 = (h16*)carve((size_t)320 * HD * 2);
  h16*   pW3_1  = (h16*)carve((size_t)HD * ED * 2);
  h16*   pW3_2  = (h16*)carve((size_t)HD * ED * 2);
  h16*   pWhh1  = (h16*)carve((size_t)192 * HD * 2);
  h16*   pWhh2  = (h16*)carve((size_t)192 * HD * 2);
  float* bias1  = (float*)carve((size_t)320 * 4);
  float* bias2  = (float*)carve((size_t)320 * 4);

  // ---- weight repacks (f32 -> f16, N x K row-major) ----
  auto pk = [&](const float* src, int rs, int co, h16* dst, int cnt, int K) {
    pack_f16_k<<<(cnt + 255) / 256, 256, 0, stream>>>(src, rs, co, dst, cnt, K);
  };
  pk(Wm1, 160, 0,   packA1,            64 * 64,  64);   // W1
  pk(Wm1, 160, 64,  packA1 + 64 * 64,  64 * 64,  64);   // W2
  pk(Wih1, 64, 0,   packA1 + 128 * 64, 192 * 64, 64);   // Wih
  pk(Wm1, 160, 128, pW3_1,             64 * 32,  32);   // W3
  pk(Whh1, 64, 0,   pWhh1,             192 * 64, 64);
  pk(Wm2, 160, 0,   packA2,            64 * 64,  64);
  pk(Wm2, 160, 64,  packA2 + 64 * 64,  64 * 64,  64);
  pk(Wih2, 64, 0,   packA2 + 128 * 64, 192 * 64, 64);
  pk(Wm2, 160, 128, pW3_2,             64 * 32,  32);
  pk(Whh2, 64, 0,   pWhh2,             192 * 64, 64);
  bias320_k<<<2, 256, 0, stream>>>(bih1, bias1);
  bias320_k<<<2, 256, 0, stream>>>(bih2, bias2);

  // ---- ehat / deg: single streaming pass over e (shared by both layers) ----
  ehat_deg_k<<<RTOT, 256, 0, stream>>>(e, adj, ehat, deg);

  // ---- two MPNN layers ----
  run_layer(h,  h1, adj, bm1, bhh1, packA1, bias1, pW3_1, pWhh1,
            ehat, deg, P, Q, S, mb, gh, t1T, stream);
  run_layer(h1, h2, adj, bm2, bhh2, packA2, bias2, pW3_2, pWhh2,
            ehat, deg, P, Q, S, mb, gh, t1T, stream);

  // ---- residual + passthrough outputs (h+h2, e, adj) ----
  add_k<<<RTOT * HD / 256, 256, 0, stream>>>(h, h2, out);
  hipMemcpyAsync(out + RTOT * HD, e,
                 (size_t)NB * NN * NN * ED * sizeof(float),
                 hipMemcpyDeviceToDevice, stream);
  hipMemcpyAsync(out + RTOT * HD + (size_t)NB * NN * NN * ED, adj,
                 (size_t)NB * NN * NN * sizeof(float),
                 hipMemcpyDeviceToDevice, stream);
}
